// MulTemporalKplanesOffsetfields_3298534884027
// MI455X (gfx1250) — compile-verified, matrix operand order checked by reading, not run on hardware
//
#include <hip/hip_runtime.h>

typedef __attribute__((ext_vector_type(16))) _Float16 v16h;
typedef __attribute__((ext_vector_type(8)))  float    v8f;

#define TILE 64          // points per block
#define WPTS 16          // points per wave (WMMA M)
#define FSTRIDE 136      // featLDS row stride in halves (16B-aligned rows, 4-bank skew)
#define HSTRIDE 65       // hLDS row stride in floats (bank skew)
#define WSTRIDE 136      // w1T row stride in halves

#define HW_T (150 * 128) // temporal plane H*W
#define HW_S (128 * 128) // spatial  plane H*W

struct SampleParams { int off[4]; float wx, wy; };

// ---- relayout [C, H*W] f32 -> [H*W, C] f16 (planes are L2-resident; coalesced writes) ----
__global__ __launch_bounds__(256) void relayout_plane(const float* __restrict__ src,
                                                      void* __restrict__ dstv, int hw)
{
  _Float16* dst = (_Float16*)dstv;
  int idx = blockIdx.x * 256 + threadIdx.x;  // out index = s*64 + c
  if (idx < hw * 64) {
    int c = idx & 63;
    int s = idx >> 6;
    dst[idx] = (_Float16)src[c * hw + s];
  }
}

template <bool F16P>
__global__ __launch_bounds__(128) void kplanes_wmma_kernel(
    const float* __restrict__ inp, const float* __restrict__ aabb,
    const void* __restrict__ pxt, const void* __restrict__ pyt,
    const void* __restrict__ pzt, const void* __restrict__ pxy,
    const void* __restrict__ pxz, const void* __restrict__ pyz,
    const float* __restrict__ w1, const float* __restrict__ b1,
    const float* __restrict__ w2, const float* __restrict__ b2,
    float* __restrict__ out, int n)
{
  // spar (phases 0-1) overlays hLDS (phases 2-3): 9216 B <= 16640 B
  __shared__ __align__(16) char smem0[TILE * HSTRIDE * 4];
  __shared__ _Float16 featLDS[TILE * FSTRIDE];   // 17408 B
  __shared__ _Float16 w1T[64 * WSTRIDE];         // 17408 B  [n][k] f16
  SampleParams (*spar)[6] = (SampleParams (*)[6])smem0;
  float* hLDS = (float*)smem0;

  const int tid  = threadIdx.x;
  const int lane = tid & 31;
  const int wave = tid >> 5;
  const int base = blockIdx.x * TILE;

  // ---- stage w1 (f32, [k][n], k*64+n linear) into LDS as f16 [n][k] ----
  for (int i = tid; i < 128 * 64; i += 128) {
    int k = i >> 6, nn = i & 63;
    w1T[nn * WSTRIDE + k] = (_Float16)w1[i];
  }

  // ---- Phase 0: per-point bilinear params (threads 0..63) ----
  if (tid < TILE) {
    int p = base + tid;
    float px = 0.f, py = 0.f, pz = 0.f, pt = 0.f;
    if (p < n) {
      float4 v4 = *(const float4*)(inp + (size_t)p * 4);
      px = v4.x; py = v4.y; pz = v4.z; pt = v4.w;
    }
    float lo0 = aabb[0], lo1 = aabb[1], lo2 = aabb[2];
    float hi0 = aabb[3], hi1 = aabb[4], hi2 = aabb[5];
    float x = (px - lo0) / (hi0 - lo0) * 2.f - 1.f;
    float y = (py - lo1) / (hi1 - lo1) * 2.f - 1.f;
    float z = (pz - lo2) / (hi2 - lo2) * 2.f - 1.f;
    float t = pt * 2.f - 1.f;
    // samples: 0:tp_xt(x,t) 1:tp_yt(y,t) 2:tp_zt(z,t) 3:sp_xy(x,y) 4:sp_xz(x,z) 5:sp_yz(y,z)
    const float us[6] = { x, y, z, x, x, y };
    const float vs[6] = { t, t, t, y, z, z };
    const int   Hs[6] = { 150, 150, 150, 128, 128, 128 };
    #pragma unroll
    for (int s = 0; s < 6; ++s) {
      const int W = 128, H = Hs[s];
      float fx = (us[s] + 1.f) * 0.5f * (float)(W - 1);
      float fy = (vs[s] + 1.f) * 0.5f * (float)(H - 1);
      float x0f = fminf(fmaxf(floorf(fx), 0.f), (float)(W - 1));
      float y0f = fminf(fmaxf(floorf(fy), 0.f), (float)(H - 1));
      int x0 = (int)x0f, y0 = (int)y0f;
      int x1 = min(x0 + 1, W - 1), y1 = min(y0 + 1, H - 1);
      spar[tid][s].off[0] = y0 * W + x0;
      spar[tid][s].off[1] = y0 * W + x1;
      spar[tid][s].off[2] = y1 * W + x0;
      spar[tid][s].off[3] = y1 * W + x1;
      spar[tid][s].wx = fx - x0f;
      spar[tid][s].wy = fy - y0f;
    }
  }
  __syncthreads();

  // ---- Phase 1: gather features; thread owns channel c = tid for all 64 points ----
  {
    const int  c        = tid;                 // 0..127
    const bool temporal = (c < 64);
    const int  ch       = temporal ? c : c - 64;
    const int  sb       = temporal ? 0 : 3;

    if (F16P) {
      // planes are [H*W, 64] f16; pre-offset by channel; coalesced across lanes
      const _Float16* pl0 = (const _Float16*)(temporal ? pxt : pxy) + ch;
      const _Float16* pl1 = (const _Float16*)(temporal ? pyt : pxz) + ch;
      const _Float16* pl2 = (const _Float16*)(temporal ? pzt : pyz) + ch;
      for (int p = 0; p < TILE; ++p) {
        float acc = 0.f;
        #pragma unroll
        for (int s = 0; s < 3; ++s) {
          const SampleParams& P = spar[p][sb + s];
          const _Float16* pl = (s == 0) ? pl0 : (s == 1) ? pl1 : pl2;
          float wx = P.wx, wy = P.wy;
          float f00 = (float)pl[(size_t)P.off[0] * 64];
          float f01 = (float)pl[(size_t)P.off[1] * 64];
          float f10 = (float)pl[(size_t)P.off[2] * 64];
          float f11 = (float)pl[(size_t)P.off[3] * 64];
          acc += f00 * (1.f - wx) * (1.f - wy) + f01 * wx * (1.f - wy)
               + f10 * (1.f - wx) * wy        + f11 * wx * wy;
        }
        featLDS[p * FSTRIDE + c] = (_Float16)acc;
      }
    } else {
      // fallback: original [64, H*W] f32 layout
      const int HW = temporal ? HW_T : HW_S;
      const float* pl0 = (const float*)(temporal ? pxt : pxy) + (size_t)ch * HW;
      const float* pl1 = (const float*)(temporal ? pyt : pxz) + (size_t)ch * HW;
      const float* pl2 = (const float*)(temporal ? pzt : pyz) + (size_t)ch * HW;
      for (int p = 0; p < TILE; ++p) {
        float acc = 0.f;
        #pragma unroll
        for (int s = 0; s < 3; ++s) {
          const SampleParams& P = spar[p][sb + s];
          const float* pl = (s == 0) ? pl0 : (s == 1) ? pl1 : pl2;
          float wx = P.wx, wy = P.wy;
          float f00 = pl[P.off[0]], f01 = pl[P.off[1]];
          float f10 = pl[P.off[2]], f11 = pl[P.off[3]];
          acc += f00 * (1.f - wx) * (1.f - wy) + f01 * wx * (1.f - wy)
               + f10 * (1.f - wx) * wy        + f11 * wx * wy;
        }
        featLDS[p * FSTRIDE + c] = (_Float16)acc;
      }
    }
  }
  __syncthreads();

  // ---- Phase 2: layer 1 via WMMA; each wave owns 16 points ----
  const int prow = wave * WPTS;
  v8f acc[4] = {};
  {
    const int rl   = lane & 15;
    const int half = lane >> 4;
    #pragma unroll
    for (int kt = 0; kt < 4; ++kt) {
      // A fragment: feat rows; element e<8: k=kt*32+half*8+e ; e>=8: k=kt*32+16+half*8+(e-8)
      v16h afrag;
      {
        const _Float16* src = &featLDS[(prow + rl) * FSTRIDE + kt * 32 + half * 8];
        uint4* ap = (uint4*)&afrag;
        ap[0] = *(const uint4*)(src);
        ap[1] = *(const uint4*)(src + 16);
      }
      #pragma unroll
      for (int nt = 0; nt < 4; ++nt) {
        // B fragment from w1T [n][k]: lane holds column n = nt*16 + rl, same k runs as A
        v16h bfrag;
        {
          const _Float16* src = &w1T[(nt * 16 + rl) * WSTRIDE + kt * 32 + half * 8];
          uint4* bp = (uint4*)&bfrag;
          bp[0] = *(const uint4*)(src);
          bp[1] = *(const uint4*)(src + 16);
        }
        acc[nt] = __builtin_amdgcn_wmma_f32_16x16x32_f16(
            false, afrag, false, bfrag, (short)0, acc[nt], false, false);
      }
    }
  }
  // bias + ReLU, scatter h to LDS (C layout: VGPR v -> M = v + 8*half, lane -> N)
  {
    const int half = lane >> 4;
    #pragma unroll
    for (int nt = 0; nt < 4; ++nt) {
      int ncol = nt * 16 + (lane & 15);
      float bias = b1[ncol];
      float* hp = (float*)&acc[nt];
      #pragma unroll
      for (int v = 0; v < 8; ++v) {
        int m = v + 8 * half;
        hLDS[(prow + m) * HSTRIDE + ncol] = fmaxf(hp[v] + bias, 0.f);
      }
    }
  }
  __syncthreads();

  // ---- Phase 3: layer 2 (64 -> 3), lanes 0..15 per wave ----
  if (lane < WPTS) {
    int p = base + prow + lane;
    if (p < n) {
      float o0 = b2[0], o1 = b2[1], o2 = b2[2];
      const float* hrow = &hLDS[(prow + lane) * HSTRIDE];
      #pragma unroll 8
      for (int k2 = 0; k2 < 64; ++k2) {
        float hv = hrow[k2];
        o0 += hv * w2[k2 * 3 + 0];
        o1 += hv * w2[k2 * 3 + 1];
        o2 += hv * w2[k2 * 3 + 2];
      }
      out[(size_t)p * 3 + 0] = o0;
      out[(size_t)p * 3 + 1] = o1;
      out[(size_t)p * 3 + 2] = o2;
    }
  }
}

extern "C" void kernel_launch(void* const* d_in, const int* in_sizes, int n_in,
                              void* d_out, int out_size, void* d_ws, size_t ws_size,
                              hipStream_t stream)
{
  const float* inp   = (const float*)d_in[0];
  const float* aabb  = (const float*)d_in[1];
  const float* tp_xt = (const float*)d_in[2];
  const float* tp_yt = (const float*)d_in[3];
  const float* tp_zt = (const float*)d_in[4];
  const float* sp_xy = (const float*)d_in[5];
  const float* sp_xz = (const float*)d_in[6];
  const float* sp_yz = (const float*)d_in[7];
  const float* w1    = (const float*)d_in[8];
  const float* b1    = (const float*)d_in[9];
  const float* w2    = (const float*)d_in[10];
  const float* b2    = (const float*)d_in[11];
  float* outp = (float*)d_out;
  int n = in_sizes[0] / 4;
  int blocks = (n + TILE - 1) / TILE;

  const size_t elemsT = (size_t)64 * HW_T;   // per temporal plane
  const size_t elemsS = (size_t)64 * HW_S;   // per spatial plane
  const size_t need   = (3 * elemsT + 3 * elemsS) * 2;  // f16 bytes

  if (ws_size >= need) {
    unsigned short* ws = (unsigned short*)d_ws;
    unsigned short* q_xt = ws;
    unsigned short* q_yt = q_xt + elemsT;
    unsigned short* q_zt = q_yt + elemsT;
    unsigned short* q_xy = q_zt + elemsT;
    unsigned short* q_xz = q_xy + elemsS;
    unsigned short* q_yz = q_xz + elemsS;
    int gT = (int)((elemsT + 255) / 256), gS = (int)((elemsS + 255) / 256);
    relayout_plane<<<gT, 256, 0, stream>>>(tp_xt, q_xt, HW_T);
    relayout_plane<<<gT, 256, 0, stream>>>(tp_yt, q_yt, HW_T);
    relayout_plane<<<gT, 256, 0, stream>>>(tp_zt, q_zt, HW_T);
    relayout_plane<<<gS, 256, 0, stream>>>(sp_xy, q_xy, HW_S);
    relayout_plane<<<gS, 256, 0, stream>>>(sp_xz, q_xz, HW_S);
    relayout_plane<<<gS, 256, 0, stream>>>(sp_yz, q_yz, HW_S);
    kplanes_wmma_kernel<true><<<blocks, 128, 0, stream>>>(
        inp, aabb, q_xt, q_yt, q_zt, q_xy, q_xz, q_yz,
        w1, b1, w2, b2, outp, n);
  } else {
    kplanes_wmma_kernel<false><<<blocks, 128, 0, stream>>>(
        inp, aabb, tp_xt, tp_yt, tp_zt, sp_xy, sp_xz, sp_yz,
        w1, b1, w2, b2, outp, n);
  }
}
